// WindowAttention_5884105195917
// MI455X (gfx1250) — compile-verified, matrix operand order checked by reading, not run on hardware
//
#include <hip/hip_runtime.h>

#define NTOK   49
#define NPAD   64
#define HEADS  12
#define HD     32
#define HIDDEN 384
#define QKVDIM 1152
#define NWIN   4096
#define TABLE  169

typedef __attribute__((ext_vector_type(16))) __bf16 v16bf;
typedef __attribute__((ext_vector_type(8)))  __bf16 v8bf;
typedef __attribute__((ext_vector_type(16))) float  v16f;
typedef __attribute__((ext_vector_type(8)))  float  v8f;

union Frag16 {
    v16bf  bf;
    v8bf   h[2];
    __bf16 e[16];
};

__global__ __launch_bounds__(256)
void swin_window_attn_kernel(const float* __restrict__ x,
                             const float* __restrict__ bias_table,
                             const float* __restrict__ qkv_w,
                             const float* __restrict__ qkv_b,
                             const float* __restrict__ fc_w,
                             const float* __restrict__ fc_b,
                             const int*   __restrict__ rel_index,
                             float*       __restrict__ out)
{
    // ---- LDS (≈274 KB of the 320 KB/WGP budget) ----
    __shared__ __align__(16) __bf16 lds_q [NPAD * HIDDEN];   // 48 KB, q pre-scaled, row-major [tok][hd]
    __shared__ __align__(16) __bf16 lds_k [NPAD * HIDDEN];   // 48 KB, row-major [tok][hd]
    __shared__ __align__(16) __bf16 lds_vT[HIDDEN * NPAD];   // 48 KB, transposed [hd][tok]
    __shared__ __align__(16) __bf16 lds_xo[NPAD * HIDDEN];   // 48 KB: x (ph1), attn-out (ph2/3)
    __shared__ __align__(16) __bf16 lds_p [8][NPAD * NPAD];  // 64 KB: per-wave softmax P tile
    __shared__ float lds_bias[TABLE * HEADS];                // 8.1 KB
    __shared__ int   lds_idx[NTOK * NTOK];                   // 9.6 KB

    const int tid  = threadIdx.x;
    const int lane = tid & 31;
    const int wv   = tid >> 5;      // 8 waves
    const int ln   = lane & 15;     // column within 16-wide tile
    const int hf   = lane >> 4;     // lane half (WMMA 16-bit K split)
    const int win  = blockIdx.x;
    const float scale = 0.17677669529663689f;   // 32^-0.5

    const float* xw = x + (size_t)win * NTOK * HIDDEN;

    // ---- Stage x (zero-pad rows 49..63), bias table, rel index ----
    for (int i = tid; i < NPAD * HIDDEN; i += 256) {
        int r = i / HIDDEN, c = i - r * HIDDEN;
        float v = (r < NTOK) ? xw[r * HIDDEN + c] : 0.0f;
        lds_xo[i] = (__bf16)v;
    }
    for (int i = tid; i < TABLE * HEADS; i += 256) lds_bias[i] = bias_table[i];
    for (int i = tid; i < NTOK * NTOK;  i += 256) lds_idx[i]  = rel_index[i];
    __syncthreads();

    // ================= Phase 1: QKV = X @ qkv_w^T + qkv_b =================
    // 72 N-tiles of 16 over 8 waves; K = 384 in 12 steps of 32.
    // Output de-interleaved: col = h*96 + d*3 + s  ->  plane s, feature hd = col/3.
    for (int nt = wv * 9; nt < wv * 9 + 9; ++nt) {
        const int col = nt * 16 + ln;            // output feature (0..1151)
        const int s   = col % 3;                 // 0:q 1:k 2:v  (per-lane constant)
        const int hd  = col / 3;                 // h*32 + d
        v8f acc[4] = {};
        for (int ks = 0; ks < 12; ++ks) {
            const int k0 = ks * 32;
            // B frag: B[k][col] = qkv_w[col*384 + k]  (fp32 -> bf16 packed convert)
            Frag16 bfr;
            const float* wp = qkv_w + (size_t)col * HIDDEN + k0 + hf * 8;
            __builtin_prefetch(wp + 32, 0, 0);
            v16f wf;
            #pragma unroll
            for (int j = 0; j < 8; ++j) { wf[j] = wp[j]; wf[8 + j] = wp[16 + j]; }
            bfr.bf = __builtin_convertvector(wf, v16bf);
            #pragma unroll
            for (int mt = 0; mt < 4; ++mt) {
                Frag16 afr;
                const __bf16* ap = &lds_xo[(mt * 16 + ln) * HIDDEN + k0 + hf * 8];
                afr.h[0] = *(const v8bf*)(ap);
                afr.h[1] = *(const v8bf*)(ap + 16);
                acc[mt] = __builtin_amdgcn_wmma_f32_16x16x32_bf16(
                    false, afr.bf, false, bfr.bf, (short)0, acc[mt], false, false);
            }
        }
        const float qb = qkv_b[col];
        // per-lane destination plane (col is fixed per lane)
        __bf16* plane = (s == 0) ? lds_q : (s == 1) ? lds_k : lds_vT;
        #pragma unroll
        for (int mt = 0; mt < 4; ++mt)
            #pragma unroll
            for (int r = 0; r < 8; ++r) {
                int m = mt * 16 + hf * 8 + r;
                float v = acc[mt][r] + qb;
                if (s == 0) v *= scale;                       // fold q scaling (matches reference)
                int addr = (s == 2) ? hd * NPAD + m : m * HIDDEN + hd;
                plane[addr] = (__bf16)v;
            }
    }
    __syncthreads();

    // ================= Phase 2: per-head attention =================
    for (int h = wv; h < HEADS; h += 8) {
        const int hb = h * HD;
        __bf16* P = lds_p[wv];

        for (int mt = 0; mt < 4; ++mt) {
            // A frag: q rows (K = head dim = 32, one WMMA step) -- contiguous
            Frag16 qf;
            {
                const __bf16* qp = &lds_q[(mt * 16 + ln) * HIDDEN + hb + hf * 8];
                qf.h[0] = *(const v8bf*)(qp);
                qf.h[1] = *(const v8bf*)(qp + 16);
            }
            v8f s4[4];
            #pragma unroll
            for (int nt = 0; nt < 4; ++nt) {
                // B frag: k^T -> lane = token, regs = d -- contiguous in k plane
                Frag16 kf;
                const __bf16* kp = &lds_k[(nt * 16 + ln) * HIDDEN + hb + hf * 8];
                kf.h[0] = *(const v8bf*)(kp);
                kf.h[1] = *(const v8bf*)(kp + 16);
                v8f z = {};
                s4[nt] = __builtin_amdgcn_wmma_f32_16x16x32_bf16(
                    false, qf.bf, false, kf.bf, (short)0, z, false, false);
            }
            // ---- softmax on these 16 rows (row lives in 16 lanes of one VGPR) ----
            #pragma unroll
            for (int r = 0; r < 8; ++r) {
                const int m = mt * 16 + hf * 8 + r;
                float mx = -3.0e38f;
                #pragma unroll
                for (int nt = 0; nt < 4; ++nt) {
                    int n = nt * 16 + ln;
                    float sv = s4[nt][r];
                    if (m < NTOK && n < NTOK)
                        sv += lds_bias[lds_idx[m * NTOK + n] * HEADS + h];
                    if (n >= NTOK) sv = -3.0e38f;       // mask pad columns
                    s4[nt][r] = sv;
                    mx = fmaxf(mx, sv);
                }
                #pragma unroll
                for (int o = 1; o < 16; o <<= 1) mx = fmaxf(mx, __shfl_xor(mx, o, 16));
                float sum = 0.f;
                #pragma unroll
                for (int nt = 0; nt < 4; ++nt) {
                    float e = __expf(s4[nt][r] - mx);
                    s4[nt][r] = e;
                    sum += e;
                }
                #pragma unroll
                for (int o = 1; o < 16; o <<= 1) sum += __shfl_xor(sum, o, 16);
                float rinv = 1.0f / sum;
                #pragma unroll
                for (int nt = 0; nt < 4; ++nt)
                    P[m * NPAD + nt * 16 + ln] = (__bf16)(s4[nt][r] * rinv);
            }
        }

        // ---- O = P @ V  (K = 64 tokens in 2 steps) -> lds_xo[:, h*32 : h*32+32]
        for (int mt = 0; mt < 4; ++mt) {
            v8f oacc[2] = {};
            #pragma unroll
            for (int kk = 0; kk < 2; ++kk) {
                Frag16 pf;   // A frag from P: lane = out row, regs = token -- contiguous
                const __bf16* pp = &P[(mt * 16 + ln) * NPAD + kk * 32 + hf * 8];
                pf.h[0] = *(const v8bf*)(pp);
                pf.h[1] = *(const v8bf*)(pp + 16);
                #pragma unroll
                for (int nt = 0; nt < 2; ++nt) {
                    // B frag: lane = d, regs = token -- contiguous in vT plane
                    Frag16 vf;
                    const __bf16* vp = &lds_vT[(hb + nt * 16 + ln) * NPAD + kk * 32 + hf * 8];
                    vf.h[0] = *(const v8bf*)(vp);
                    vf.h[1] = *(const v8bf*)(vp + 16);
                    oacc[nt] = __builtin_amdgcn_wmma_f32_16x16x32_bf16(
                        false, pf.bf, false, vf.bf, (short)0, oacc[nt], false, false);
                }
            }
            #pragma unroll
            for (int nt = 0; nt < 2; ++nt)
                #pragma unroll
                for (int r = 0; r < 8; ++r) {
                    int m = mt * 16 + hf * 8 + r;
                    lds_xo[m * HIDDEN + hb + nt * 16 + ln] = (__bf16)oacc[nt][r];
                }
        }
    }
    __syncthreads();

    // ================= Phase 3: out = O @ fc_w^T + fc_b =================
    float* ow = out + (size_t)win * NTOK * HIDDEN;
    for (int nt = wv * 3; nt < wv * 3 + 3; ++nt) {
        const int col = nt * 16 + ln;
        v8f acc[4] = {};
        for (int ks = 0; ks < 12; ++ks) {
            const int k0 = ks * 32;
            Frag16 bfr;
            const float* wp = fc_w + (size_t)col * HIDDEN + k0 + hf * 8;
            __builtin_prefetch(wp + 32, 0, 0);
            v16f wf;
            #pragma unroll
            for (int j = 0; j < 8; ++j) { wf[j] = wp[j]; wf[8 + j] = wp[16 + j]; }
            bfr.bf = __builtin_convertvector(wf, v16bf);
            #pragma unroll
            for (int mt = 0; mt < 4; ++mt) {
                Frag16 afr;
                const __bf16* ap = &lds_xo[(mt * 16 + ln) * HIDDEN + k0 + hf * 8];
                afr.h[0] = *(const v8bf*)(ap);
                afr.h[1] = *(const v8bf*)(ap + 16);
                acc[mt] = __builtin_amdgcn_wmma_f32_16x16x32_bf16(
                    false, afr.bf, false, bfr.bf, (short)0, acc[mt], false, false);
            }
        }
        const float fb = fc_b[col];
        #pragma unroll
        for (int mt = 0; mt < 4; ++mt)
            #pragma unroll
            for (int r = 0; r < 8; ++r) {
                int m = mt * 16 + hf * 8 + r;
                if (m < NTOK) ow[m * HIDDEN + col] = acc[mt][r] + fb;
            }
    }
}

extern "C" void kernel_launch(void* const* d_in, const int* in_sizes, int n_in,
                              void* d_out, int out_size, void* d_ws, size_t ws_size,
                              hipStream_t stream) {
    const float* x    = (const float*)d_in[0];
    const float* tab  = (const float*)d_in[1];
    const float* qw   = (const float*)d_in[2];
    const float* qb   = (const float*)d_in[3];
    const float* fw   = (const float*)d_in[4];
    const float* fb   = (const float*)d_in[5];
    const int*   ridx = (const int*)d_in[6];
    swin_window_attn_kernel<<<NWIN, 256, 0, stream>>>(
        x, tab, qw, qb, fw, fb, ridx, (float*)d_out);
}
